// GAE_48378511622556
// MI455X (gfx1250) — compile-verified, matrix operand order checked by reading, not run on hardware
//
#include <hip/hip_runtime.h>
#include <hip/hip_bf16.h>
#include <math.h>

// ---------------------------------------------------------------------------
// CDNA5 (gfx1250) implementation of the 2-level GCN reference.
// All [*,128] x [128,128] (and one [*,256]x[256,128]) matmuls run on
// v_wmma_f32_16x16x32_bf16 (bf16 A/B, f32 accumulate). Weight tiles are
// staged into LDS with gfx1250 async global->LDS copies (ASYNCcnt).
// Graph scatter/gather stays on VALU + f32 atomics (bandwidth-bound).
// ---------------------------------------------------------------------------

typedef __attribute__((ext_vector_type(16))) __bf16 bf16x16;
typedef __attribute__((ext_vector_type(8)))  float  f32x8;

#define H 128
#define NTILES 8   // 128 output cols / 16

__device__ __forceinline__ float elu_f(float v) {
  return v > 0.0f ? v : expm1f(v);
}

// gfx1250 async copy: 16 bytes global -> LDS, tracked by ASYNCcnt.
// GVS addressing: mem = s[base] + v_goff; vdst = LDS byte address.
__device__ __forceinline__ void async_ld16(unsigned ldsOff, const void* base,
                                           unsigned goff) {
  asm volatile("global_load_async_to_lds_b128 %0, %1, %2"
               :: "v"(ldsOff), "v"(goff), "s"(base)
               : "memory");
}
__device__ __forceinline__ void wait_async0() {
  asm volatile("s_wait_asynccnt 0" ::: "memory");
}

// A-fragment loader: 16x32 bf16 tile, M = lane%16, K packing per ISA 7.12.2:
// K = (half<8 ? 0 : 16) + (half%8) + (lane>=16 ? 8 : 0)
// p points at row base + kTile*32 + (lane>=16 ? 8 : 0), row-major f32 source.
__device__ __forceinline__ bf16x16 load_a_frag_f32(const float* __restrict__ p) {
  bf16x16 a;
#pragma unroll
  for (int h = 0; h < 8; ++h) a[h] = (__bf16)p[h];
#pragma unroll
  for (int h = 0; h < 8; ++h) a[8 + h] = (__bf16)p[16 + h];
  return a;
}

// ---------------------------------------------------------------------------
// Weight prep: f32 row-major [K,128] -> bf16 WMMA B-fragments.
// Fragment tile (kT,nT): 32 lanes x 16 halves; N = lane%16,
// K = (half<8?0:16) + half%8 + (lane>=16?8:0) within the 32-K tile.
// ---------------------------------------------------------------------------
__global__ void prep_weight_kernel(const float* __restrict__ W,
                                   __bf16* __restrict__ Wp, int K) {
  int idx = blockIdx.x * 256 + threadIdx.x;
  if (idx >= K * H) return;
  int k = idx >> 7, n = idx & 127;
  int kT = k >> 5, kk = k & 31, nT = n >> 4, nn = n & 15;
  int lane = nn | (((kk >> 3) & 1) << 4);
  int h    = (kk & 7) | ((kk >> 4) << 3);
  Wp[((((kT << 3) | nT) * 32 + lane) << 4) | h] = (__bf16)W[idx];
}

// ---------------------------------------------------------------------------
// Generic GEMM: Y[N,128] = act(X[N,KT*32] @ W + b)
// 256 threads = 8 waves, each wave computes a 16x128 strip via 8 WMMA tiles.
// Weights stream into LDS via async global->LDS while each wave preloads
// ALL its A-fragments into registers; then the K loop is pure
// (16x ds_load_b128 -> 8x v_wmma) per K-tile, back-to-back.
// ---------------------------------------------------------------------------
template <int KT, bool BIAS, bool ELU>
__global__ __launch_bounds__(256) void gemm_kernel(
    const float* __restrict__ X, const __bf16* __restrict__ Wp,
    const float* __restrict__ bias, float* __restrict__ Y, int N) {
  extern __shared__ char smem[];
  __bf16* ldsW = (__bf16*)smem;                       // KT*32*128 bf16
  const int tid = threadIdx.x, lane = tid & 31, wv = tid >> 5;
  const unsigned ldsBase = __builtin_amdgcn_groupstaticsize();

  // async-stage the prepped weights into LDS (ASYNCcnt path)
  const int nv4 = KT * 32 * 16;                       // uint4 count
#pragma unroll 2
  for (int i = tid; i < nv4; i += 256)
    async_ld16(ldsBase + (unsigned)i * 16u, Wp, (unsigned)i * 16u);

  const int m   = lane & 15;
  const int hi8 = (lane >> 4) << 3;
  const int rowBase = blockIdx.x * 128 + (wv << 4);
  int row = rowBase + m;
  if (row > N - 1) row = N - 1;                       // clamp: keep EXEC all-1s
  const float* xrow = X + (size_t)row * (KT * 32);

  // preload all A-fragments (overlaps with async weight staging)
  bf16x16 afr[KT];
#pragma unroll
  for (int kT = 0; kT < KT; ++kT)
    afr[kT] = load_a_frag_f32(xrow + (kT << 5) + hi8);

  wait_async0();
  __syncthreads();

  f32x8 acc[NTILES] = {};
#pragma unroll
  for (int kT = 0; kT < KT; ++kT) {
    bf16x16 bfr[NTILES];
#pragma unroll
    for (int nT = 0; nT < NTILES; ++nT)
      bfr[nT] = *(const bf16x16*)(ldsW + ((((kT << 3) | nT) * 32 + lane) << 4));
#pragma unroll
    for (int nT = 0; nT < NTILES; ++nT)
      acc[nT] = __builtin_amdgcn_wmma_f32_16x16x32_bf16(
          false, afr[kT], false, bfr[nT], (short)0, acc[nT], false, false);
  }

#pragma unroll
  for (int nT = 0; nT < NTILES; ++nT) {
    const int col = (nT << 4) | m;
    const float bc = BIAS ? bias[col] : 0.0f;
#pragma unroll
    for (int j = 0; j < 8; ++j) {                      // C: VGPR j -> M = j(+8)
      const int r = rowBase + j + hi8;
      if (r < N) {
        float v = acc[nT][j] + bc;
        if (ELU) v = elu_f(v);
        Y[(size_t)r * H + col] = v;
      }
    }
  }
}

// ---------------------------------------------------------------------------
// Fused edge decoder: w[e] = ELU( ELU( ELU(e*W0+b0) @ W1 + b1 ) . W2 + b2 )
// 128 edges per block. Layer-0 output is written straight into LDS in
// A-fragment layout; layer-1 is WMMA; layer-2 is a shuffle-reduced dot.
// LDS: 32KB A-fragments + 32KB W1 fragments (W1 staged asynchronously).
// ---------------------------------------------------------------------------
__global__ __launch_bounds__(256) void edge_dec_kernel(
    const float* __restrict__ ea, const float* __restrict__ W0,
    const float* __restrict__ b0, const __bf16* __restrict__ W1p,
    const float* __restrict__ b1, const float* __restrict__ W2,
    const float* __restrict__ b2, float* __restrict__ wout, int E) {
  extern __shared__ char smem[];
  __bf16* ldsA = (__bf16*)smem;                 // 128x128 A-fragments (32 KB)
  __bf16* ldsW = (__bf16*)(smem + 32768);       // W1 fragments        (32 KB)
  const int tid = threadIdx.x, lane = tid & 31, wv = tid >> 5;
  const int eBase = blockIdx.x * 128;
  const unsigned ldsBase = __builtin_amdgcn_groupstaticsize();

  // async-stage W1 fragments while we compute layer 0
#pragma unroll 2
  for (int i = tid; i < H * 16; i += 256)       // 2048 uint4 = 32 KB
    async_ld16(ldsBase + 32768u + (unsigned)i * 16u, W1p, (unsigned)i * 16u);

  for (int idx = tid; idx < H * H; idx += 256) {
    int i = idx >> 7, j = idx & 127;            // i: edge-in-block, j: hidden
    int ei = eBase + i;
    if (ei > E - 1) ei = E - 1;
    float v = elu_f(ea[ei] * W0[j] + b0[j]);
    int mT = i >> 4, mm = i & 15, kT = j >> 5, kk = j & 31;
    int ln = mm | (((kk >> 3) & 1) << 4);
    int h  = (kk & 7) | ((kk >> 4) << 3);
    ldsA[((((mT << 2) | kT) * 32 + ln) << 4) | h] = (__bf16)v;
  }
  wait_async0();
  __syncthreads();

  f32x8 acc[NTILES] = {};
#pragma unroll
  for (int kT = 0; kT < 4; ++kT) {
    bf16x16 a = *(const bf16x16*)(ldsA + ((((wv << 2) | kT) * 32 + lane) << 4));
    bf16x16 bfr[NTILES];
#pragma unroll
    for (int nT = 0; nT < NTILES; ++nT)
      bfr[nT] = *(const bf16x16*)(ldsW + ((((kT << 3) | nT) * 32 + lane) << 4));
#pragma unroll
    for (int nT = 0; nT < NTILES; ++nT)
      acc[nT] = __builtin_amdgcn_wmma_f32_16x16x32_bf16(
          false, a, false, bfr[nT], (short)0, acc[nT], false, false);
  }

  // layer 2: per-row dot with W2[128] (ELU after layer 1 fused in)
  const int m = lane & 15;
  float part[8];
#pragma unroll
  for (int j = 0; j < 8; ++j) part[j] = 0.0f;
#pragma unroll
  for (int nT = 0; nT < NTILES; ++nT) {
    const int col = (nT << 4) | m;
    const float bc = b1[col], wc = W2[col];
#pragma unroll
    for (int j = 0; j < 8; ++j) part[j] += elu_f(acc[nT][j] + bc) * wc;
  }
#pragma unroll
  for (int msk = 1; msk < 16; msk <<= 1) {      // reduce within 16-lane halves
#pragma unroll
    for (int j = 0; j < 8; ++j) part[j] += __shfl_xor(part[j], msk, 32);
  }
  if (m == 0) {
    const float b2v = b2[0];
    const int r0 = eBase + (wv << 4) + ((lane >> 4) << 3);
#pragma unroll
    for (int j = 0; j < 8; ++j) {
      int ei = r0 + j;
      if (ei < E) wout[ei] = elu_f(part[j] + b2v);
    }
  }
}

// ---------------------------------------------------------------------------
// GCN helper kernels
// ---------------------------------------------------------------------------
__global__ void fill_kernel(float* __restrict__ p, float v, int n) {
  int i = blockIdx.x * 256 + threadIdx.x;
  if (i < n) p[i] = v;
}

__global__ void deg_scatter_kernel(const int* __restrict__ col,
                                   const float* __restrict__ w,
                                   float* __restrict__ deg, int E) {
  int e = blockIdx.x * 256 + threadIdx.x;
  if (e < E) atomicAdd(&deg[col[e]], w[e]);
}

__global__ void dinv_kernel(const float* __restrict__ deg,
                            float* __restrict__ dinv, int n) {
  int i = blockIdx.x * 256 + threadIdx.x;
  if (i >= n) return;
  float d = deg[i];
  dinv[i] = d > 0.0f ? rsqrtf(fmaxf(d, 1e-12f)) : 0.0f;
}

__global__ void self_init_kernel(const float* __restrict__ hw,
                                 const float* __restrict__ dinv,
                                 float* __restrict__ out, int n) {
  int idx = blockIdx.x * 256 + threadIdx.x;
  if (idx >= n * H) return;
  float di = dinv[idx >> 7];
  out[idx] = hw[idx] * di * di;                 // self-loop (w=1) contribution
}

__global__ __launch_bounds__(256) void msg_scatter_kernel(
    const int* __restrict__ rowI, const int* __restrict__ colI,
    const float* __restrict__ w, const float* __restrict__ dinv,
    const float* __restrict__ hw, float* __restrict__ out, int E) {
  int e = blockIdx.x * 8 + (threadIdx.x >> 5);
  int lane = threadIdx.x & 31;
  if (e >= E) return;
  int r = rowI[e], c = colI[e];
  float nrm = dinv[r] * w[e] * dinv[c];
  float4 hv = ((const float4*)(hw + (size_t)r * H))[lane];
  float* op = out + (size_t)c * H + lane * 4;
  atomicAdd(op + 0, hv.x * nrm);
  atomicAdd(op + 1, hv.y * nrm);
  atomicAdd(op + 2, hv.z * nrm);
  atomicAdd(op + 3, hv.w * nrm);
}

__global__ void bias_elu_kernel(float* __restrict__ x,
                                const float* __restrict__ b, int n) {
  int idx = blockIdx.x * 256 + threadIdx.x;
  if (idx >= n * H) return;
  x[idx] = elu_f(x[idx] + b[idx & 127]);
}

// ---------------------------------------------------------------------------
// c2f layer 0: out = ELU((pos_c[cl[i]]-pos_f[i]) @ W[2,128] + b)
// ---------------------------------------------------------------------------
__global__ void c2f0_kernel(const float* __restrict__ posf,
                            const float* __restrict__ posc,
                            const int* __restrict__ cl,
                            const float* __restrict__ W,
                            const float* __restrict__ b,
                            float* __restrict__ out, int n) {
  int idx = blockIdx.x * 256 + threadIdx.x;
  if (idx >= n * H) return;
  int i = idx >> 7, j = idx & 127;
  int c = cl[i];
  float d0 = posc[c * 2 + 0] - posf[i * 2 + 0];
  float d1 = posc[c * 2 + 1] - posf[i * 2 + 1];
  out[idx] = elu_f(d0 * W[j] + d1 * W[H + j] + b[j]);
}

__global__ void concat_kernel(const float* __restrict__ ea,
                              const float* __restrict__ xc,
                              const int* __restrict__ cl,
                              float* __restrict__ t, int n) {
  int idx = blockIdx.x * 256 + threadIdx.x;
  if (idx >= n * 256) return;
  int i = idx >> 8, j = idx & 255;
  t[idx] = (j < H) ? ea[(size_t)i * H + j]
                   : xc[(size_t)cl[i] * H + (j - H)];
}

// residual + LayerNorm, one wave per 128-wide row
__global__ __launch_bounds__(256) void res_ln_kernel(
    const float* __restrict__ ea, const float* __restrict__ t,
    const float* __restrict__ g, const float* __restrict__ bb,
    float* __restrict__ out, int n) {
  int r = blockIdx.x * 8 + (threadIdx.x >> 5);
  int lane = threadIdx.x & 31;
  if (r >= n) return;
  float4 tv = ((const float4*)(t + (size_t)r * H))[lane];
  float4 ev = ((const float4*)(ea + (size_t)r * H))[lane];
  float4 v = {tv.x + ev.x, tv.y + ev.y, tv.z + ev.z, tv.w + ev.w};
  float s  = v.x + v.y + v.z + v.w;
  float s2 = v.x * v.x + v.y * v.y + v.z * v.z + v.w * v.w;
#pragma unroll
  for (int msk = 1; msk < 32; msk <<= 1) {
    s  += __shfl_xor(s, msk, 32);
    s2 += __shfl_xor(s2, msk, 32);
  }
  float mu  = s * (1.0f / H);
  float var = s2 * (1.0f / H) - mu * mu;
  float rs  = rsqrtf(var + 1e-5f);
  int c = lane * 4;
  float4 o;
  o.x = (v.x - mu) * rs * g[c + 0] + bb[c + 0];
  o.y = (v.y - mu) * rs * g[c + 1] + bb[c + 1];
  o.z = (v.z - mu) * rs * g[c + 2] + bb[c + 2];
  o.w = (v.w - mu) * rs * g[c + 3] + bb[c + 3];
  ((float4*)(out + (size_t)r * H))[lane] = o;
}

// node head: y[r] = ELU(X[r,:] . w + b), one wave per row
__global__ __launch_bounds__(256) void rowdot_elu_kernel(
    const float* __restrict__ X, const float* __restrict__ w,
    const float* __restrict__ b, float* __restrict__ y, int n) {
  int r = blockIdx.x * 8 + (threadIdx.x >> 5);
  int lane = threadIdx.x & 31;
  if (r >= n) return;
  float4 xv = ((const float4*)(X + (size_t)r * H))[lane];
  float4 wv = ((const float4*)w)[lane];
  float s = xv.x * wv.x + xv.y * wv.y + xv.z * wv.z + xv.w * wv.w;
#pragma unroll
  for (int msk = 1; msk < 32; msk <<= 1) s += __shfl_xor(s, msk, 32);
  if (lane == 0) y[r] = elu_f(s + b[0]);
}

__global__ void conv1d_kernel(const float* __restrict__ x,
                              const float* __restrict__ cw,
                              const float* __restrict__ cb,
                              float* __restrict__ y, int n) {
  int i = blockIdx.x * 256 + threadIdx.x;
  if (i >= n) return;
  float a = (i > 0) ? x[i - 1] : 0.0f;
  float m = x[i];
  float c = (i + 1 < n) ? x[i + 1] : 0.0f;
  y[i] = cw[0] * a + cw[1] * m + cw[2] * c + cb[0];
}

// ---------------------------------------------------------------------------
// Host orchestration
// ---------------------------------------------------------------------------
static inline int cdiv_i(int a, int b) { return (a + b - 1) / b; }

extern "C" void kernel_launch(void* const* d_in, const int* in_sizes, int n_in,
                              void* d_out, int out_size, void* d_ws, size_t ws_size,
                              hipStream_t stream) {
  (void)in_sizes; (void)n_in; (void)out_size; (void)ws_size;
  const int NC = 25000, NF = 100000, EC = 200000, EF = 800000;

  // ---- inputs, in setup_inputs() / _make_params() insertion order ----
  int ix = 0;
  const float* x_in   = (const float*)d_in[ix++];
  const float* eattrC = (const float*)d_in[ix++];
  const float* eattrF = (const float*)d_in[ix++];
  const float* posf   = (const float*)d_in[ix++];
  const float* posc   = (const float*)d_in[ix++];
  const float *edW[2][3], *edB[2][3], *gW[2][2], *gB[2][2];
  const float *c2fW[3], *c2fB[3], *upW[3], *upB[3], *ndW[3], *ndB[3];
  for (int b = 0; b < 2; ++b) for (int l = 0; l < 3; ++l) edW[b][l] = (const float*)d_in[ix++];
  for (int b = 0; b < 2; ++b) for (int l = 0; l < 3; ++l) edB[b][l] = (const float*)d_in[ix++];
  for (int b = 0; b < 2; ++b) for (int l = 0; l < 2; ++l) gW[b][l]  = (const float*)d_in[ix++];
  for (int b = 0; b < 2; ++b) for (int l = 0; l < 2; ++l) gB[b][l]  = (const float*)d_in[ix++];
  for (int l = 0; l < 3; ++l) c2fW[l] = (const float*)d_in[ix++];
  for (int l = 0; l < 3; ++l) c2fB[l] = (const float*)d_in[ix++];
  for (int l = 0; l < 3; ++l) upW[l]  = (const float*)d_in[ix++];
  for (int l = 0; l < 3; ++l) upB[l]  = (const float*)d_in[ix++];
  for (int l = 0; l < 3; ++l) ndW[l]  = (const float*)d_in[ix++];
  for (int l = 0; l < 3; ++l) ndB[l]  = (const float*)d_in[ix++];
  const float* lng = (const float*)d_in[ix++];
  const float* lnb = (const float*)d_in[ix++];
  const float* cw  = (const float*)d_in[ix++];
  const float* cb  = (const float*)d_in[ix++];
  const int* eiC = (const int*)d_in[ix++];   // [2, EC]
  const int* eiF = (const int*)d_in[ix++];   // [2, EF]
  const int* cl  = (const int*)d_in[ix++];   // [NF]

  // ---- workspace carve-up ----
  size_t off = 0;
  auto carve = [&](size_t bytes) -> char* {
    char* p = (char*)d_ws + off;
    off += (bytes + 255) & ~(size_t)255;
    return p;
  };
  float* B0 = (float*)carve((size_t)NC * H * 4);
  float* B1 = (float*)carve((size_t)NC * H * 4);
  float* B2 = (float*)carve((size_t)NC * H * 4);
  float* F0 = (float*)carve((size_t)NF * H * 4);
  float* F1 = (float*)carve((size_t)NF * H * 4);
  float* F2 = (float*)carve((size_t)NF * H * 4);
  float* F3 = (float*)carve((size_t)NF * 256 * 4);
  float* w1 = (float*)carve((size_t)EC * 4);
  float* w0 = (float*)carve((size_t)EF * 4);
  float* deg  = (float*)carve((size_t)NF * 4);
  float* dinv = (float*)carve((size_t)NF * 4);
  float* nodev = (float*)carve((size_t)NF * 4);
  const size_t WB = (size_t)H * H * sizeof(__bf16);   // 32 KB per 128x128
  __bf16* pEd[2]   = {(__bf16*)carve(WB), (__bf16*)carve(WB)};
  __bf16* pG[2][2] = {{(__bf16*)carve(WB), (__bf16*)carve(WB)},
                      {(__bf16*)carve(WB), (__bf16*)carve(WB)}};
  __bf16* pC2f[2]  = {(__bf16*)carve(WB), (__bf16*)carve(WB)};
  __bf16* pUp0     = (__bf16*)carve(2 * WB);          // 256x128
  __bf16* pUp[2]   = {(__bf16*)carve(WB), (__bf16*)carve(WB)};
  __bf16* pNd[2]   = {(__bf16*)carve(WB), (__bf16*)carve(WB)};

  dim3 blk(256);
  auto prep = [&](const float* W, __bf16* Wp, int K) {
    prep_weight_kernel<<<cdiv_i(K * H, 256), blk, 0, stream>>>(W, Wp, K);
  };
  prep(edW[0][1], pEd[0], H);   prep(edW[1][1], pEd[1], H);
  prep(gW[0][0], pG[0][0], H);  prep(gW[0][1], pG[0][1], H);
  prep(gW[1][0], pG[1][0], H);  prep(gW[1][1], pG[1][1], H);
  prep(c2fW[1], pC2f[0], H);    prep(c2fW[2], pC2f[1], H);
  prep(upW[0], pUp0, 256);      prep(upW[1], pUp[0], H);  prep(upW[2], pUp[1], H);
  prep(ndW[0], pNd[0], H);      prep(ndW[1], pNd[1], H);

  // K = 128 (KT=4) or 256 (KT=8); bias+ELU fused when requested
  auto gemm = [&](const float* X, const __bf16* Wp, const float* bias, float* Y,
                  int N, int K, bool be) {
    dim3 g(cdiv_i(N, 128));
    size_t sh = (size_t)K * 256;                      // K*128 bf16
    if (K == 256) {
      gemm_kernel<8, true, true><<<g, blk, sh, stream>>>(X, Wp, bias, Y, N);
    } else if (be) {
      gemm_kernel<4, true, true><<<g, blk, sh, stream>>>(X, Wp, bias, Y, N);
    } else {
      gemm_kernel<4, false, false><<<g, blk, sh, stream>>>(X, Wp, bias, Y, N);
    }
  };

  auto gcn_layer = [&](const float* xin, float* hw, float* xout, const __bf16* Wp,
                       const float* bias, const int* rowI, const int* colI,
                       const float* wE, int n, int E) {
    gemm(xin, Wp, nullptr, hw, n, H, false);
    fill_kernel<<<cdiv_i(n, 256), blk, 0, stream>>>(deg, 1.0f, n);
    deg_scatter_kernel<<<cdiv_i(E, 256), blk, 0, stream>>>(colI, wE, deg, E);
    dinv_kernel<<<cdiv_i(n, 256), blk, 0, stream>>>(deg, dinv, n);
    self_init_kernel<<<cdiv_i(n * H, 256), blk, 0, stream>>>(hw, dinv, xout, n);
    msg_scatter_kernel<<<cdiv_i(E, 8), blk, 0, stream>>>(rowI, colI, wE, dinv, hw, xout, E);
    bias_elu_kernel<<<cdiv_i(n * H, 256), blk, 0, stream>>>(xout, bias, n);
  };

  const size_t edSh = 65536;                          // 32KB A + 32KB W1

  // ---- block 0: coarse ----
  edge_dec_kernel<<<cdiv_i(EC, 128), blk, edSh, stream>>>(
      eattrC, edW[0][0], edB[0][0], pEd[0], edB[0][1], edW[0][2], edB[0][2], w1, EC);
  gcn_layer(x_in, B2, B0, pG[0][0], gB[0][0], eiC, eiC + EC, w1, NC, EC);
  gcn_layer(B0,   B2, B1, pG[0][1], gB[0][1], eiC, eiC + EC, w1, NC, EC);

  // ---- coarse -> fine refinement ----
  c2f0_kernel<<<cdiv_i(NF * H, 256), blk, 0, stream>>>(posf, posc, cl, c2fW[0], c2fB[0], F1, NF);
  gemm(F1, pC2f[0], c2fB[1], F2, NF, H, true);
  gemm(F2, pC2f[1], c2fB[2], F0, NF, H, true);        // F0 = ea_c2f
  concat_kernel<<<cdiv_i(NF * 256, 256), blk, 0, stream>>>(F0, B1, cl, F3, NF);
  gemm(F3, pUp0, upB[0], F1, NF, 256, true);
  gemm(F1, pUp[0], upB[1], F2, NF, H, true);
  gemm(F2, pUp[1], upB[2], F1, NF, H, true);          // t = F1
  res_ln_kernel<<<cdiv_i(NF, 8), blk, 0, stream>>>(F0, F1, lng, lnb, F2, NF);  // x = F2

  // ---- block 1: fine ----
  edge_dec_kernel<<<cdiv_i(EF, 128), blk, edSh, stream>>>(
      eattrF, edW[1][0], edB[1][0], pEd[1], edB[1][1], edW[1][2], edB[1][2], w0, EF);
  gcn_layer(F2, F3, F1, pG[1][0], gB[1][0], eiF, eiF + EF, w0, NF, EF);
  gcn_layer(F1, F3, F2, pG[1][1], gB[1][1], eiF, eiF + EF, w0, NF, EF);

  // ---- node decoder + conv1d ----
  gemm(F2, pNd[0], ndB[0], F1, NF, H, true);
  gemm(F1, pNd[1], ndB[1], F3, NF, H, true);          // F3 reused as [NF,128]
  rowdot_elu_kernel<<<cdiv_i(NF, 8), blk, 0, stream>>>(F3, ndW[2], ndB[2], nodev, NF);
  conv1d_kernel<<<cdiv_i(NF, 256), blk, 0, stream>>>(nodev, cw, cb, (float*)d_out, NF);
}
